// TrellisMoELayer_678604833228
// MI455X (gfx1250) — matrix operand order repaired and checked by reading, not compiled, not run on hardware
//
#include <hip/hip_runtime.h>
#include <hip/hip_bf16.h>

#define T_DIM 2048
#define D_DIM 1024
#define F_DIM 2816
#define E_DIM 8

typedef __attribute__((ext_vector_type(16))) __bf16 v16bf;
typedef __attribute__((ext_vector_type(8)))  float  v8f;
typedef __attribute__((ext_vector_type(4)))  float  v4f;
typedef __attribute__((ext_vector_type(2)))  unsigned int v2u;

union BF16Frag { v16bf v; unsigned int u[8]; };

#if defined(__has_builtin)
#if __has_builtin(__builtin_amdgcn_perm)
#define HAVE_PERM 1
#endif
#endif

// round-to-nearest (ties away) bf16; 2 ops
__device__ __forceinline__ unsigned short f2bf(float f) {
  return (unsigned short)((__float_as_uint(f) + 0x8000u) >> 16);
}

// pack two fp32 -> packed bf16 dword: 2x v_add + 1x v_perm
__device__ __forceinline__ unsigned int pack2bf(float a, float b) {
  unsigned int ua = __float_as_uint(a) + 0x8000u;
  unsigned int ub = __float_as_uint(b) + 0x8000u;
#ifdef HAVE_PERM
  // D.b0=S1.b2 D.b1=S1.b3 D.b2=S0.b2 D.b3=S0.b3  (S0=ub, S1=ua)
  return __builtin_amdgcn_perm(ub, ua, 0x07060302u);
#else
  return (ua >> 16) | (ub & 0xffff0000u);
#endif
}

// Fragment-order LDS addressing (per CDNA5 ISA §7.12.2, wave32).
// A (16x32 bf16), this code's contiguous K order (B matches): lane<16 holds K 0..15; lane>=16: K 16..31.
//   For K%4==0 the quad (K..K+3) maps to one lane, dwords v, v+1 (v even).
__device__ __forceinline__ int a_slot(int row, int K_even, int sub /*16-row subtile*/) {
  int hi = K_even >> 4;
  int v = (K_even & 15) >> 1;
  return (((sub << 5) + (hi << 4) + row) << 3) + v;   // uint index
}
// B (32x16 bf16): lane = 16*(K>>4) + N; dword idx = (K&15)/2.
// STRIDE16 variant: each (subtile,lane) owns 16 dwords (two interleaved fragments).
__device__ __forceinline__ int b_slot16(int N, int K_even, int sub) {
  int lane = ((K_even >> 4) << 4) + N;
  return (((sub << 5) + lane) << 4) + ((K_even & 15) >> 1);
}
__device__ __forceinline__ int b_slot8(int N, int K_even, int sub) {
  int lane = ((K_even >> 4) << 4) + N;
  return (((sub << 5) + lane) << 3) + ((K_even & 15) >> 1);
}

// ---------------------------------------------------------------- zero
__global__ __launch_bounds__(256) void k_zero(float* __restrict__ out, int n4,
                                              int* __restrict__ counts) {
  int i = blockIdx.x * 256 + threadIdx.x;
  if (i < n4) {
    v4f z = {0.f, 0.f, 0.f, 0.f};
    *(v4f*)&out[(size_t)i * 4] = z;
  }
  if (i < E_DIM) counts[i] = 0;
}

// ---------------------------------------------------------------- router
__global__ __launch_bounds__(256) void k_router(
    const float* __restrict__ x, const float* __restrict__ Wr,
    int* __restrict__ counts, int* __restrict__ tok_id, float* __restrict__ tok_w) {
  const int lane = threadIdx.x & 31;
  const int wave = threadIdx.x >> 5;
  const int t = blockIdx.x * 8 + wave;
  __shared__ float slog[8][8];
  for (int e = 0; e < E_DIM; ++e) {
    float acc = 0.f;
#pragma unroll
    for (int j = 0; j < D_DIM / 128; ++j) {
      int d = (j * 32 + lane) * 4;
      v4f xv = *(const v4f*)&x[(size_t)t * D_DIM + d];
      v4f wv = *(const v4f*)&Wr[(size_t)e * D_DIM + d];
      acc += xv.x * wv.x + xv.y * wv.y + xv.z * wv.z + xv.w * wv.w;
    }
    for (int off = 16; off > 0; off >>= 1)
      acc += __shfl_xor(acc, off, 32);
    if (lane == 0) slog[wave][e] = acc;
  }
  if (lane == 0) {
    float l0 = -3.4e38f, l1 = -3.4e38f; int e0 = 0, e1 = 0;
    for (int e = 0; e < E_DIM; ++e) {
      float l = slog[wave][e];
      if (l > l0) { l1 = l0; e1 = e0; l0 = l; e0 = e; }
      else if (l > l1) { l1 = l; e1 = e; }
    }
    float w0 = 1.f / (1.f + __expf(l1 - l0));
    float w1 = 1.f - w0;
    int s0 = atomicAdd(&counts[e0], 1);
    tok_id[e0 * T_DIM + s0] = t;  tok_w[e0 * T_DIM + s0] = w0;
    int s1 = atomicAdd(&counts[e1], 1);
    tok_id[e1 * T_DIM + s1] = t;  tok_w[e1 * T_DIM + s1] = w1;
  }
}

// ---------------------------------------------------------------- prefix
__global__ void k_prefix(const int* __restrict__ counts, int* __restrict__ base) {
  if (threadIdx.x == 0) {
    int s = 0;
    for (int e = 0; e < E_DIM; ++e) { base[e] = s; s += counts[e]; }
  }
}

// ---------------------------------------------------------------- gate/up
__global__ __launch_bounds__(256) void k_gateup(
    const float* __restrict__ x, const float* __restrict__ Wg,
    const float* __restrict__ Wu, const int* __restrict__ counts,
    const int* __restrict__ base, const int* __restrict__ tok_id,
    unsigned short* __restrict__ hbuf) {
  const int e   = blockIdx.z;
  const int cnt = counts[e];
  const int m0  = blockIdx.x * 64;
  if (m0 >= cnt) return;
  const int f0  = blockIdx.y * 128;

  __shared__ unsigned int sTok[64];
  __shared__ unsigned int sAu[64 * 8];      // 4 subtiles x 32 lanes x 8 dwords
  __shared__ unsigned int sBgu[128 * 16];   // 8 subtiles x 32 lanes x {8 gate, 8 up}

  const int tix = threadIdx.x;
  if (tix < 64) {
    // clamp instead of sentinel: garbage rows only affect masked output rows,
    // and the unconditional load keeps staging vectorized (global_load_b128)
    int r = min(m0 + tix, cnt - 1);
    sTok[tix] = (unsigned int)tok_id[e * T_DIM + r];
  }

  const int lane = tix & 31;
  const int wave = tix >> 5;
  const int mi = wave & 3;      // 16-row sub-tile
  const int nh = wave >> 2;     // 64-col half
  const int hi = lane >> 4;
  const int lo = lane & 15;

  v8f zero8 = {0.f, 0.f, 0.f, 0.f, 0.f, 0.f, 0.f, 0.f};
  v8f accg[4], accu[4];
#pragma unroll
  for (int j = 0; j < 4; ++j) { accg[j] = zero8; accu[j] = zero8; }

  for (int k0 = 0; k0 < D_DIM; k0 += 32) {
    __syncthreads();
    // A: 64 tokens x 32 d -> v4f load, pack, ds_store_b64
#pragma unroll
    for (int p = 0; p < 2; ++p) {
      int qi = tix + p * 256;            // quad index 0..511
      int r = qi >> 3;                   // token row 0..63
      int c = (qi & 7) << 2;             // K multiple of 4
      v4f xv = *(const v4f*)&x[(size_t)sTok[r] * D_DIM + k0 + c];
      v2u pk = {pack2bf(xv.x, xv.y), pack2bf(xv.z, xv.w)};
      *(v2u*)&sAu[a_slot(r & 15, c, r >> 4)] = pk;
    }
    // B: 128 f x 32 d from W_gate/W_up, interleaved fragments in one array
#pragma unroll
    for (int p = 0; p < 4; ++p) {
      int qi = tix + p * 256;            // quad index 0..1023
      int fr = qi >> 3;                  // f row 0..127
      int c = (qi & 7) << 2;             // K (d), multiple of 4
      size_t off = ((size_t)e * F_DIM + f0 + fr) * D_DIM + k0 + c;
      v4f g4 = *(const v4f*)&Wg[off];
      v4f u4 = *(const v4f*)&Wu[off];
      if ((qi & 7) == 0) {               // speculative prefetch next K-chunk
        __builtin_prefetch(&Wg[off + 32], 0, 0);
        __builtin_prefetch(&Wu[off + 32], 0, 0);
      }
      int slot = b_slot16(fr & 15, c, fr >> 4);
      v2u pg = {pack2bf(g4.x, g4.y), pack2bf(g4.z, g4.w)};
      v2u pu = {pack2bf(u4.x, u4.y), pack2bf(u4.z, u4.w)};
      *(v2u*)&sBgu[slot]     = pg;
      *(v2u*)&sBgu[slot + 8] = pu;
    }
    __syncthreads();

    BF16Frag af;
    const unsigned int* ap = &sAu[((mi << 5) + lane) << 3];
#pragma unroll
    for (int v = 0; v < 8; ++v) af.u[v] = ap[v];

#pragma unroll
    for (int j = 0; j < 4; ++j) {
      BF16Frag bg, bu;
      const unsigned int* bp = &sBgu[(((nh * 4 + j) << 5) + lane) << 4];
#pragma unroll
      for (int v = 0; v < 8; ++v) { bg.u[v] = bp[v]; bu.u[v] = bp[8 + v]; }
      accg[j] = __builtin_amdgcn_wmma_f32_16x16x32_bf16(
          false, af.v, false, bg.v, (short)0, accg[j], false, false);
      accu[j] = __builtin_amdgcn_wmma_f32_16x16x32_bf16(
          false, af.v, false, bu.v, (short)0, accu[j], false, false);
    }
  }

  // epilogue: h = silu(g) * u, store bf16 to compacted h buffer
  const int be = base[e];
#pragma unroll
  for (int j = 0; j < 4; ++j) {
#pragma unroll
    for (int r = 0; r < 8; ++r) {
      int lrow = mi * 16 + r + hi * 8;
      int grow = m0 + lrow;
      if (grow < cnt) {
        float g = accg[j][r];
        float u = accu[j][r];
        float h = (g / (1.f + __expf(-g))) * u;
        hbuf[(size_t)(be + grow) * F_DIM + (f0 + nh * 64 + j * 16 + lo)] = f2bf(h);
      }
    }
  }
}

// ---------------------------------------------------------------- down + combine
__global__ __launch_bounds__(256) void k_down(
    const unsigned short* __restrict__ hbuf, const float* __restrict__ Wd,
    const int* __restrict__ counts, const int* __restrict__ base,
    const int* __restrict__ tok_id, const float* __restrict__ tok_w,
    float* __restrict__ out) {
  const int e   = blockIdx.z;
  const int cnt = counts[e];
  const int m0  = blockIdx.x * 64;
  if (m0 >= cnt) return;
  const int d0  = blockIdx.y * 128;
  const int be  = base[e];

  __shared__ unsigned int sTok[64];
  __shared__ float sW[64];
  __shared__ unsigned int sAu[64 * 8];
  __shared__ unsigned int sB[128 * 8];

  const int tix = threadIdx.x;
  if (tix < 64) {
    int r = min(m0 + tix, cnt - 1);
    sTok[tix] = (unsigned int)tok_id[e * T_DIM + r];
    sW[tix]   = tok_w[e * T_DIM + r];
  }

  const int lane = tix & 31;
  const int wave = tix >> 5;
  const int mi = wave & 3;
  const int nh = wave >> 2;
  const int hi = lane >> 4;
  const int lo = lane & 15;

  v8f zero8 = {0.f, 0.f, 0.f, 0.f, 0.f, 0.f, 0.f, 0.f};
  v8f acc[4];
#pragma unroll
  for (int j = 0; j < 4; ++j) acc[j] = zero8;

  for (int k0 = 0; k0 < F_DIM; k0 += 32) {
    __syncthreads();
    // A: h rows, already bf16 -> unconditional v2u load (clamped row)
#pragma unroll
    for (int p = 0; p < 2; ++p) {
      int qi = tix + p * 256;
      int r = qi >> 3;
      int c = (qi & 7) << 2;
      int grow = min(m0 + r, cnt - 1);
      v2u pk = *(const v2u*)&hbuf[(size_t)(be + grow) * F_DIM + k0 + c];
      *(v2u*)&sAu[a_slot(r & 15, c, r >> 4)] = pk;
    }
    // B: W_down[e] is [D, F] row-major over f
#pragma unroll
    for (int p = 0; p < 4; ++p) {
      int qi = tix + p * 256;
      int dr = qi >> 3;                  // d row 0..127
      int c = (qi & 7) << 2;             // K (f), multiple of 4
      size_t off = ((size_t)e * D_DIM + d0 + dr) * F_DIM + k0 + c;
      v4f w4 = *(const v4f*)&Wd[off];
      if ((qi & 7) == 0)
        __builtin_prefetch(&Wd[off + 32], 0, 0);
      v2u pw = {pack2bf(w4.x, w4.y), pack2bf(w4.z, w4.w)};
      *(v2u*)&sB[b_slot8(dr & 15, c, dr >> 4)] = pw;
    }
    __syncthreads();

    BF16Frag af;
    const unsigned int* ap = &sAu[((mi << 5) + lane) << 3];
#pragma unroll
    for (int v = 0; v < 8; ++v) af.u[v] = ap[v];

#pragma unroll
    for (int j = 0; j < 4; ++j) {
      BF16Frag bf;
      const unsigned int* bp = &sB[(((nh * 4 + j) << 5) + lane) << 3];
#pragma unroll
      for (int v = 0; v < 8; ++v) bf.u[v] = bp[v];
      acc[j] = __builtin_amdgcn_wmma_f32_16x16x32_bf16(
          false, af.v, false, bf.v, (short)0, acc[j], false, false);
    }
  }

  // epilogue: out[tok, d] += weight * y   (two experts per token -> atomic)
#pragma unroll
  for (int j = 0; j < 4; ++j) {
#pragma unroll
    for (int r = 0; r < 8; ++r) {
      int lrow = mi * 16 + r + hi * 8;
      int grow = m0 + lrow;
      if (grow < cnt) {
        unsigned int tok = sTok[lrow];
        float w = sW[lrow];
        atomicAdd(&out[(size_t)tok * D_DIM + (d0 + nh * 64 + j * 16 + lo)],
                  w * acc[j][r]);
      }
    }
  }
}

// ---------------------------------------------------------------- launch
extern "C" void kernel_launch(void* const* d_in, const int* in_sizes, int n_in,
                              void* d_out, int out_size, void* d_ws, size_t ws_size,
                              hipStream_t stream) {
  (void)in_sizes; (void)n_in; (void)out_size; (void)ws_size;
  const float* x  = (const float*)d_in[0];
  const float* Wr = (const float*)d_in[1];
  const float* Wg = (const float*)d_in[2];
  const float* Wu = (const float*)d_in[3];
  const float* Wd = (const float*)d_in[4];
  float* out = (float*)d_out;

  // workspace layout
  char* ws = (char*)d_ws;
  int*   counts = (int*)ws;                                        // 8 ints
  int*   base   = counts + 8;                                      // 8 ints
  int*   tok_id = (int*)(ws + 64);                                 // E*T ints (64 KB)
  float* tok_w  = (float*)(ws + 64 + E_DIM * T_DIM * 4);           // 64 KB
  unsigned short* hbuf = (unsigned short*)(ws + 131328);           // 2T x F bf16 (~22 MB)

  const int n4 = T_DIM * D_DIM / 4;
  k_zero<<<(n4 + 255) / 256, 256, 0, stream>>>(out, n4, counts);
  k_router<<<T_DIM / 8, 256, 0, stream>>>(x, Wr, counts, tok_id, tok_w);
  k_prefix<<<1, 32, 0, stream>>>(counts, base);
  k_gateup<<<dim3(T_DIM / 64, F_DIM / 128, E_DIM), 256, 0, stream>>>(
      x, Wg, Wu, counts, base, tok_id, hbuf);
  k_down<<<dim3(T_DIM / 64, D_DIM / 128, E_DIM), 256, 0, stream>>>(
      hbuf, Wd, counts, base, tok_id, tok_w, out);
}
